// Model1_63745904608086
// MI455X (gfx1250) — matrix operand order repaired and checked by reading, not compiled, not run on hardware
//
#include <hip/hip_runtime.h>
#include <math.h>

#define NTOT        100000
#define IN_CHN      1024
#define MDIM        512
#define ATT_D       128
#define NUM_CLS     2
#define NUM_GROUP   8
#define INST_PG     4
#define GROUP_SIZE  12500
#define MPAD        12544   // 49 * 256, instance dim padded to WMMA tiling

typedef __attribute__((ext_vector_type(16))) _Float16 v16h;
typedef __attribute__((ext_vector_type(8)))  float    v8f;

// ---------------------------------------------------------------------------
// Fragment loaders: convert fp32 memory to f16 WMMA fragments in registers.
// A-fragment (16x32, MxK): this lane supplies 16 contiguous K values at
// [p..p+15] (caller applies the lane-half +16 K offset).
// ---------------------------------------------------------------------------
__device__ inline v16h load_fragA_f32(const float* __restrict__ p) {
    const float4* q = (const float4*)p;
    float4 x0 = q[0], x1 = q[1], y0 = q[2], y1 = q[3];
    v16h v;
    v[0]=(_Float16)x0.x; v[1]=(_Float16)x0.y; v[2]=(_Float16)x0.z; v[3]=(_Float16)x0.w;
    v[4]=(_Float16)x1.x; v[5]=(_Float16)x1.y; v[6]=(_Float16)x1.z; v[7]=(_Float16)x1.w;
    v[8]=(_Float16)y0.x; v[9]=(_Float16)y0.y; v[10]=(_Float16)y0.z; v[11]=(_Float16)y0.w;
    v[12]=(_Float16)y1.x; v[13]=(_Float16)y1.y; v[14]=(_Float16)y1.z; v[15]=(_Float16)y1.w;
    return v;
}
// B-fragment (32x16, KxN): this lane supplies 16 contiguous K values of one
// column (caller applies the lane-half +16 K offset).
__device__ inline v16h load_fragB_f32(const float* __restrict__ p) {
    const float4* q = (const float4*)p;
    float4 x0 = q[0], x1 = q[1], x2 = q[2], x3 = q[3];
    v16h v;
    v[0]=(_Float16)x0.x; v[1]=(_Float16)x0.y; v[2]=(_Float16)x0.z; v[3]=(_Float16)x0.w;
    v[4]=(_Float16)x1.x; v[5]=(_Float16)x1.y; v[6]=(_Float16)x1.z; v[7]=(_Float16)x1.w;
    v[8]=(_Float16)x2.x; v[9]=(_Float16)x2.y; v[10]=(_Float16)x2.z; v[11]=(_Float16)x2.w;
    v[12]=(_Float16)x3.x; v[13]=(_Float16)x3.y; v[14]=(_Float16)x3.z; v[15]=(_Float16)x3.w;
    return v;
}

// ---------------------------------------------------------------------------
// Kernel 1: gather + mid = relu(sub @ W1^T + b1)   [per group, M=MPAD, N=512]
// Block = 256 threads = 8 waves; wave owns a 32(M) x 64(N) output tile:
// 2 A fragments + 4 B fragments feed 8 WMMAs per K-step (6 cvt_pk / WMMA).
// Block covers M=256; K loops over 1024 in steps of 32.
// ---------------------------------------------------------------------------
__global__ __launch_bounds__(256)
void k_gemm1(const int* __restrict__ idx, const float* __restrict__ tfeat,
             const float* __restrict__ W1, const float* __restrict__ b1,
             float* __restrict__ mid)
{
    const int g    = blockIdx.z;
    const int nB   = blockIdx.y * 64;
    const int wave = threadIdx.x >> 5;
    const int lane = threadIdx.x & 31;
    const int lrow = lane & 15;
    const int half = lane >> 4;
    const int mB   = blockIdx.x * 256 + wave * 32;

    const int m0  = mB + lrow;
    const int m1  = mB + 16 + lrow;
    const int gi0 = (m0 < GROUP_SIZE) ? idx[g * GROUP_SIZE + m0] : idx[g * GROUP_SIZE];
    const int gi1 = (m1 < GROUP_SIZE) ? idx[g * GROUP_SIZE + m1] : idx[g * GROUP_SIZE];
    const float* arow0 = tfeat + (long)gi0 * IN_CHN;
    const float* arow1 = tfeat + (long)gi1 * IN_CHN;
    const float* brow0 = W1 + (long)(nB +  0 + lrow) * IN_CHN;
    const float* brow1 = W1 + (long)(nB + 16 + lrow) * IN_CHN;
    const float* brow2 = W1 + (long)(nB + 32 + lrow) * IN_CHN;
    const float* brow3 = W1 + (long)(nB + 48 + lrow) * IN_CHN;

    const int aoff = half * 16;   // A lanes>=16: K 16..31
    const int boff = half * 16;   // B lanes>=16: K 16..31

    v8f c00 = {}, c01 = {}, c02 = {}, c03 = {};
    v8f c10 = {}, c11 = {}, c12 = {}, c13 = {};
    for (int k0 = 0; k0 < IN_CHN; k0 += 32) {
        v16h a0 = load_fragA_f32(arow0 + k0 + aoff);
        v16h a1 = load_fragA_f32(arow1 + k0 + aoff);
        v16h b0 = load_fragB_f32(brow0 + k0 + boff);
        v16h b1f= load_fragB_f32(brow1 + k0 + boff);
        v16h b2 = load_fragB_f32(brow2 + k0 + boff);
        v16h b3 = load_fragB_f32(brow3 + k0 + boff);
        c00 = __builtin_amdgcn_wmma_f32_16x16x32_f16(false, a0, false, b0, (short)0, c00, false, false);
        c01 = __builtin_amdgcn_wmma_f32_16x16x32_f16(false, a0, false, b1f,(short)0, c01, false, false);
        c02 = __builtin_amdgcn_wmma_f32_16x16x32_f16(false, a0, false, b2, (short)0, c02, false, false);
        c03 = __builtin_amdgcn_wmma_f32_16x16x32_f16(false, a0, false, b3, (short)0, c03, false, false);
        c10 = __builtin_amdgcn_wmma_f32_16x16x32_f16(false, a1, false, b0, (short)0, c10, false, false);
        c11 = __builtin_amdgcn_wmma_f32_16x16x32_f16(false, a1, false, b1f,(short)0, c11, false, false);
        c12 = __builtin_amdgcn_wmma_f32_16x16x32_f16(false, a1, false, b2, (short)0, c12, false, false);
        c13 = __builtin_amdgcn_wmma_f32_16x16x32_f16(false, a1, false, b3, (short)0, c13, false, false);
    }

    // Epilogue: bias + ReLU; store (pad rows too, so later WMMA reads are finite).
    const float bias0 = b1[nB +  0 + lrow];
    const float bias1 = b1[nB + 16 + lrow];
    const float bias2 = b1[nB + 32 + lrow];
    const float bias3 = b1[nB + 48 + lrow];
    #pragma unroll
    for (int r = 0; r < 8; ++r) {
        const int mrow = mB + r + half * 8;
        float* mg0 = mid + ((long)g * MPAD + mrow) * MDIM;
        float* mg1 = mg0 + 16 * MDIM;
        mg0[nB +  0 + lrow] = fmaxf(c00[r] + bias0, 0.f);
        mg0[nB + 16 + lrow] = fmaxf(c01[r] + bias1, 0.f);
        mg0[nB + 32 + lrow] = fmaxf(c02[r] + bias2, 0.f);
        mg0[nB + 48 + lrow] = fmaxf(c03[r] + bias3, 0.f);
        mg1[nB +  0 + lrow] = fmaxf(c10[r] + bias0, 0.f);
        mg1[nB + 16 + lrow] = fmaxf(c11[r] + bias1, 0.f);
        mg1[nB + 32 + lrow] = fmaxf(c12[r] + bias2, 0.f);
        mg1[nB + 48 + lrow] = fmaxf(c13[r] + bias3, 0.f);
    }
}

// ---------------------------------------------------------------------------
// Kernel 2: attention logits  l[s] = Wa2 . tanh(Wa1 mid[s] + ba1) + ba2
// WMMA over (mid[16x512] x Wa1^T[512x128]); tanh + Wa2 weighting fused in the
// C-layout epilogue; per-instance sum = 16-lane shfl_xor reduction.
// ---------------------------------------------------------------------------
__global__ __launch_bounds__(256)
void k_attn(const float* __restrict__ mid, const float* __restrict__ Wa1,
            const float* __restrict__ ba1, const float* __restrict__ Wa2,
            const float* __restrict__ ba2, float* __restrict__ logits)
{
    const int g    = blockIdx.y;
    const int wave = threadIdx.x >> 5;
    const int lane = threadIdx.x & 31;
    const int lrow = lane & 15;
    const int half = lane >> 4;
    const int mB   = (blockIdx.x * 8 + wave) * 16;

    const float* arow = mid + ((long)g * MPAD + mB + lrow) * MDIM;
    const int aoff = half * 16;
    const int boff = half * 16;

    float part[8] = {0.f,0.f,0.f,0.f,0.f,0.f,0.f,0.f};
    for (int nt = 0; nt < ATT_D / 16; ++nt) {
        const int n = nt * 16 + lrow;                 // this lane's column
        const float* brow = Wa1 + (long)n * MDIM;
        v8f c = {};
        for (int k0 = 0; k0 < MDIM; k0 += 32) {
            v16h a = load_fragA_f32(arow + k0 + aoff);
            v16h b = load_fragB_f32(brow + k0 + boff);
            c = __builtin_amdgcn_wmma_f32_16x16x32_f16(false, a, false, b, (short)0, c, false, false);
        }
        const float wn = Wa2[n];
        const float bn = ba1[n];
        #pragma unroll
        for (int r = 0; r < 8; ++r)
            part[r] += tanhf(c[r] + bn) * wn;
    }
    // Reduce over the 16 columns held by each lane half (same instance M).
    #pragma unroll
    for (int r = 0; r < 8; ++r) {
        float v = part[r];
        v += __shfl_xor(v, 1, 32);
        v += __shfl_xor(v, 2, 32);
        v += __shfl_xor(v, 4, 32);
        v += __shfl_xor(v, 8, 32);
        part[r] = v;
    }
    if (lrow == 0) {
        const int m = mB + half * 8;                  // lane0 -> M 0..7, lane16 -> M 8..15
        #pragma unroll
        for (int r = 0; r < 8; ++r)
            logits[(long)g * MPAD + m + r] = part[r] + ba2[0];
    }
}

// ---------------------------------------------------------------------------
// Kernel 3: raw[s] = mid[s] . (Wc[1]-Wc[0])   (argsort key precursor: 2-class
// softmax on att@Wc^T is monotone in a[s]*raw[s]).  One wave per instance.
// ---------------------------------------------------------------------------
__global__ __launch_bounds__(256)
void k_raw(const float* __restrict__ mid, const float* __restrict__ Wc,
           float* __restrict__ raw)
{
    const long inst = (long)blockIdx.x * 8 + (threadIdx.x >> 5);
    const int lane  = threadIdx.x & 31;
    if (inst >= (long)NUM_GROUP * MPAD) return;
    const float* mrow = mid + inst * MDIM;
    float acc = 0.f;
    for (int d = lane; d < MDIM; d += 32)
        acc += mrow[d] * (Wc[MDIM + d] - Wc[d]);
    acc += __shfl_xor(acc, 1, 32);
    acc += __shfl_xor(acc, 2, 32);
    acc += __shfl_xor(acc, 4, 32);
    acc += __shfl_xor(acc, 8, 32);
    acc += __shfl_xor(acc, 16, 32);
    if (lane == 0) raw[inst] = acc;
}

// ---------------------------------------------------------------------------
// Kernel 4: per-group softmax over GROUP_SIZE logits -> attention weights.
// ---------------------------------------------------------------------------
__global__ __launch_bounds__(1024)
void k_softmax(const float* __restrict__ logits, float* __restrict__ attw,
               float* __restrict__ stats)
{
    __shared__ float red[1024];
    const int g = blockIdx.x, t = threadIdx.x;
    const float* lg = logits + (long)g * MPAD;

    float mx = -3.4e38f;
    for (int s = t; s < GROUP_SIZE; s += 1024) mx = fmaxf(mx, lg[s]);
    red[t] = mx; __syncthreads();
    for (int o = 512; o > 0; o >>= 1) {
        if (t < o) red[t] = fmaxf(red[t], red[t + o]);
        __syncthreads();
    }
    mx = red[0]; __syncthreads();

    float sm = 0.f;
    for (int s = t; s < GROUP_SIZE; s += 1024) sm += __expf(lg[s] - mx);
    red[t] = sm; __syncthreads();
    for (int o = 512; o > 0; o >>= 1) {
        if (t < o) red[t] += red[t + o];
        __syncthreads();
    }
    const float inv = 1.0f / red[0];

    for (int s = t; s < MPAD; s += 1024)
        attw[(long)g * MPAD + s] = (s < GROUP_SIZE) ? __expf(lg[s] - mx) * inv : 0.f;
    if (t == 0) { stats[g * 2] = mx; stats[g * 2 + 1] = red[0]; }
}

// ---------------------------------------------------------------------------
// Kernel 5: bag = sum_s a[s]*mid[s];  pred = bag @ Wc^T + bc
// ---------------------------------------------------------------------------
__global__ __launch_bounds__(512)
void k_bag(const float* __restrict__ mid, const float* __restrict__ attw,
           const float* __restrict__ Wc, const float* __restrict__ bc,
           float* __restrict__ outPred)
{
    __shared__ float bag[MDIM];
    const int g = blockIdx.x, d = threadIdx.x;
    const float* mg = mid + (long)g * MPAD * MDIM;
    const float* aw = attw + (long)g * MPAD;
    float acc = 0.f;
    #pragma unroll 4
    for (int s = 0; s < GROUP_SIZE; ++s)
        acc += aw[s] * mg[(long)s * MDIM + d];
    bag[d] = acc;
    __syncthreads();
    if (d < NUM_CLS) {
        float p = bc[d];
        for (int k = 0; k < MDIM; ++k) p += bag[k] * Wc[d * MDIM + k];
        outPred[g * NUM_CLS + d] = p;
    }
}

// ---------------------------------------------------------------------------
// Kernel 6: 4x argmax + 4x argmin over key[s] = a[s]*raw[s]; emit mid rows in
// argsort order [top0..top3, rank S-4..S-1].
// ---------------------------------------------------------------------------
__global__ __launch_bounds__(256)
void k_topk(const float* __restrict__ mid, const float* __restrict__ attw,
            const float* __restrict__ raw, float* __restrict__ outFeat)
{
    __shared__ float bv[256];
    __shared__ int   bi[256];
    __shared__ int   sel[8];
    const int g = blockIdx.x, t = threadIdx.x;
    const float* aw = attw + (long)g * MPAD;
    const float* rw = raw  + (long)g * MPAD;

    for (int p = 0; p < 8; ++p) {
        const bool mx = (p < 4);
        float bestv = mx ? -3.4e38f : 3.4e38f;
        int   besti = -1;
        for (int s = t; s < GROUP_SIZE; s += 256) {
            bool taken = false;
            for (int q = 0; q < p; ++q) taken |= (sel[q] == s);
            if (taken) continue;
            const float key = aw[s] * rw[s];
            const bool better = mx ? (key > bestv) : (key < bestv);
            if (better || (key == bestv && (unsigned)s < (unsigned)besti)) {
                bestv = key; besti = s;
            }
        }
        bv[t] = bestv; bi[t] = besti;
        __syncthreads();
        for (int o = 128; o > 0; o >>= 1) {
            if (t < o) {
                const bool take = mx ? (bv[t + o] > bv[t]) : (bv[t + o] < bv[t]);
                if (take || (bv[t + o] == bv[t] && (unsigned)bi[t + o] < (unsigned)bi[t])) {
                    bv[t] = bv[t + o]; bi[t] = bi[t + o];
                }
            }
            __syncthreads();
        }
        if (t == 0) sel[p] = bi[0];
        __syncthreads();
    }
    // rows 0..3 = top1..top4 ; rows 4..7 = rank S-4..S-1 = sel[7],sel[6],sel[5],sel[4]
    for (int r = 0; r < 8; ++r) {
        const int src = (r < 4) ? sel[r] : sel[11 - r];
        const float* mrow = mid + ((long)g * MPAD + src) * MDIM;
        float* orow = outFeat + ((long)g * 8 + r) * MDIM;
        for (int d = t; d < MDIM; d += 256) orow[d] = mrow[d];
    }
}

// ---------------------------------------------------------------------------
extern "C" void kernel_launch(void* const* d_in, const int* in_sizes, int n_in,
                              void* d_out, int out_size, void* d_ws, size_t ws_size,
                              hipStream_t stream)
{
    const int*   idx   = (const int*)  d_in[0];
    const float* tfeat = (const float*)d_in[1];
    const float* W1    = (const float*)d_in[2];
    const float* b1    = (const float*)d_in[3];
    const float* Wa1   = (const float*)d_in[4];
    const float* ba1   = (const float*)d_in[5];
    const float* Wa2   = (const float*)d_in[6];
    const float* ba2   = (const float*)d_in[7];
    const float* Wc    = (const float*)d_in[8];
    const float* bc    = (const float*)d_in[9];
    float* out = (float*)d_out;

    float* ws     = (float*)d_ws;
    float* mid    = ws;                                          // 8*MPAD*512
    float* logits = mid    + (size_t)NUM_GROUP * MPAD * MDIM;    // 8*MPAD
    float* attw   = logits + (size_t)NUM_GROUP * MPAD;           // 8*MPAD
    float* raw    = attw   + (size_t)NUM_GROUP * MPAD;           // 8*MPAD
    float* stats  = raw    + (size_t)NUM_GROUP * MPAD;           // 8*2

    k_gemm1  <<<dim3(MPAD / 256, MDIM / 64, NUM_GROUP), dim3(256), 0, stream>>>(idx, tfeat, W1, b1, mid);
    k_attn   <<<dim3(MPAD / 128, NUM_GROUP),            dim3(256), 0, stream>>>(mid, Wa1, ba1, Wa2, ba2, logits);
    k_raw    <<<dim3(NUM_GROUP * MPAD / 8),             dim3(256), 0, stream>>>(mid, Wc, raw);
    k_softmax<<<dim3(NUM_GROUP),                        dim3(1024),0, stream>>>(logits, attw, stats);
    k_bag    <<<dim3(NUM_GROUP),                        dim3(512), 0, stream>>>(mid, attw, Wc, bc, out);
    k_topk   <<<dim3(NUM_GROUP),                        dim3(256), 0, stream>>>(mid, attw, raw, out + NUM_GROUP * NUM_CLS);
}
